// Graph_Actor_Model_48172353192218
// MI455X (gfx1250) — matrix operand order repaired and checked by reading, not compiled, not run on hardware
//
#include <hip/hip_runtime.h>
#include <hip/hip_bf16.h>
#include <stdint.h>

// ---------------------------------------------------------------------------
// MI455X (gfx1250) implementation.
// Binary adjacency is exact in bf16 -> both graph-propagation GEMMs
// (8192x8192 @ 8192x256) run on v_wmma_f32_16x16x32_bf16 with f32 accum.
// A tiles stream HBM->LDS via global_load_async_to_lds_b128 into an 8-deep
// LDS ring (prefetch distance 7, ASYNCcnt-pipelined) -> ~7MB of the HBM
// stream in flight across 128 blocks. B operands are pre-transposed
// (K-contiguous per column) so WMMA B fragments load straight from global/L2
// with 2x b128 per lane -- no LDS staging for B. dinv/bias/relu/mask fused
// in GEMM epilogues; V1/V2 written transposed as packed b128 stores.
// Degree computation fused into the adjacency transpose (saves a 268MB pass).
// ---------------------------------------------------------------------------

typedef __bf16 bf16x16 __attribute__((ext_vector_type(16)));
typedef float f32x8 __attribute__((ext_vector_type(8)));

union AFrag { uint4 u[2]; bf16x16 v; };

__device__ __forceinline__ unsigned short f32_to_bf16(float f) {
  unsigned int u = __float_as_uint(f);
  u += 0x7FFFu + ((u >> 16) & 1u);   // round-to-nearest-even
  return (unsigned short)(u >> 16);
}

// Two async b128 copies: global (flat addr) -> LDS, 32 bytes per lane.
// INST_OFFSET applies to both the global and LDS address (ISA 15.18.3).
__device__ __forceinline__ void async_copy_b128x2(const unsigned short* g,
                                                  unsigned ldsoff) {
  unsigned long long ga = (unsigned long long)(size_t)g;
  asm volatile(
      "global_load_async_to_lds_b128 %0, %1, off\n\t"
      "global_load_async_to_lds_b128 %0, %1, off offset:16"
      :: "v"(ldsoff), "v"(ga) : "memory");
}
__device__ __forceinline__ void wait_async0() {
  asm volatile("s_wait_asynccnt 0x0" ::: "memory");
}
__device__ __forceinline__ void wait_async12() {
  asm volatile("s_wait_asynccnt 0xc" ::: "memory");
}

// --------------------------- helper kernels --------------------------------

__global__ void zero_f32_kernel(float* __restrict__ p, int n) {
  int i = blockIdx.x * 256 + threadIdx.x;
  if (i < n) p[i] = 0.0f;
}

__global__ void dinv_kernel(const float* __restrict__ deg, float* __restrict__ dinv, int N) {
  int c = blockIdx.x * 256 + threadIdx.x;
  if (c < N) {
    float d = deg[c];
    dinv[c] = (d > 0.0f) ? rsqrtf(d) : 0.0f;
  }
}

// Fused: Bt[c, r] = (A[r, c] != 0) ? 1.0bf16 : 0  AND  deg[c] += column count.
// Tile staged as [c][r] (pad 72 -> 16B-aligned rows) so the transposed output
// is written with coalesced b128 stores. Counts are integer-valued f32 ->
// atomic order independent, exact.
__global__ void transpose_deg_kernel(const float* __restrict__ A,
                                     unsigned short* __restrict__ Bt,
                                     float* __restrict__ deg, int N) {
  __shared__ __align__(16) unsigned short tile[64][72];
  __shared__ float colsum[64];
  const int t = threadIdx.x;
  const int c0 = blockIdx.x * 64;
  const int r0 = blockIdx.y * 64;
  if (t < 64) colsum[t] = 0.0f;
  __syncthreads();
  const int c = t & 63;              // fixed column per thread (256 % 64 == 0)
  float partial = 0.0f;
  for (int i = t; i < 4096; i += 256) {
    int r = i >> 6;
    bool nz = (A[(size_t)(r0 + r) * N + (c0 + c)] != 0.0f);   // coalesced in c
    tile[c][r] = nz ? 0x3F80 : 0;
    partial += nz ? 1.0f : 0.0f;
  }
  atomicAdd(&colsum[c], partial);
  __syncthreads();
  for (int vi = t; vi < 512; vi += 256) {   // 64 cols x 8 groups of 8 rows
    int cc = vi >> 3, r8 = (vi & 7) << 3;
    *(uint4*)&Bt[(size_t)(c0 + cc) * N + r0 + r8] = *(const uint4*)&tile[cc][r8];
  }
  if (t < 64) atomicAdd(&deg[c0 + t], colsum[t]);
}

__global__ void cvt_f32_to_bf16_kernel(const float* __restrict__ src,
                                       unsigned short* __restrict__ dst, int n) {
  int i = blockIdx.x * 256 + threadIdx.x;
  if (i < n) dst[i] = f32_to_bf16(src[i]);
}

// dst[n*K + k] = bf16(src[k*Nc + n])  (transposed weights, tiny matrices)
__global__ void cvt_transpose_bf16_kernel(const float* __restrict__ src,
                                          unsigned short* __restrict__ dst,
                                          int K, int Nc) {
  int i = blockIdx.x * 256 + threadIdx.x;
  if (i < K * Nc) {
    int k = i / Nc, n = i - k * Nc;
    dst[(size_t)n * K + k] = f32_to_bf16(src[i]);
  }
}

// ----------------------------- WMMA GEMM -----------------------------------
// C[M x NCOLS] = concat_K(A1[M x K1], A2[M x K2]) @ B, with B supplied
// TRANSPOSED as BT[NCOLS][K1+K2] (K-contiguous per output column).
// Block: 64 rows x NCOLS, 8 waves: wave = (mstrip = w&3, nhalf = w>>2).
// K-step 64 (2 WMMA chunks). A tile (64x64 bf16, 8KB) streamed via
// global_load_async_to_lds_b128 into an 8-deep LDS ring, prefetch distance 7.
// B fragments load directly from global BT (L2-resident), 2x b128 per lane.
// Epilogues: 0: relu(x+b)->bf16 row-major
//            1: epi0 + outT[n][r] = bf16(dinv[r]*x)   (transposed dual)
//            2: outT[n][r] = bf16(dinv[r]^2*x)        (transposed)
//            3: bf16(dinv[r]*x) row-major
//            4: f32((x+b)*mask[r]) row-major
template <int NCOLS, int EPI>
__global__ __launch_bounds__(256) void gemm_bf16_wmma(
    const unsigned short* __restrict__ A1, int K1,
    const unsigned short* __restrict__ A2, int K2, int lda,
    const unsigned short* __restrict__ BT,
    const float* __restrict__ bias, const float* __restrict__ scale,
    unsigned short* __restrict__ out0,   // row-major bf16 [M][NCOLS]
    unsigned short* __restrict__ outT,   // transposed bf16 [NCOLS][M]
    float* __restrict__ outf, int Mrows) {
  constexpr int TILES = NCOLS / 32;      // 16-wide N tiles per wave

  __shared__ __align__(16) unsigned short As[8][64 * 64];   // 8-stage ring, 64KB

  const int t = threadIdx.x;
  const int wave = t >> 5;
  const int lane = t & 31;
  const int m0 = blockIdx.x * 64;
  const int mstrip = wave & 3;
  const int nbase = (wave >> 2) * TILES;
  const int Ktot = K1 + K2;
  const int steps = Ktot >> 6;

  const int arow = t >> 2;               // A coop-copy: 4 threads x 32B per row
  const int akc = (t & 3) << 4;

  auto issueA = [&](int p) {
    const int k0 = p << 6;
    const unsigned short* Asrc = (k0 < K1) ? A1 : A2;
    const int krel = (k0 < K1) ? k0 : (k0 - K1);
    const unsigned short* g = Asrc + (size_t)(m0 + arow) * lda + krel + akc;
    unsigned ldsoff = (unsigned)(size_t)&As[p & 7][arow * 64 + akc];
    async_copy_b128x2(g, ldsoff);
  };

  f32x8 acc[TILES];
  const f32x8 zero = {0.f, 0.f, 0.f, 0.f, 0.f, 0.f, 0.f, 0.f};
#pragma unroll
  for (int j = 0; j < TILES; ++j) acc[j] = zero;

  // prologue: fill pipeline (up to 7 stages ahead), drain, barrier.
  for (int p = 0; p < 7 && p < steps; ++p) issueA(p);
  wait_async0();
  __syncthreads();

  for (int s = 0; s < steps; ++s) {
    const unsigned short* Ast = &As[s & 7][0];
#pragma unroll
    for (int c = 0; c < 2; ++c) {
      // A frag (ISA 16-bit A 16x32): lane<16 -> K 0..7,16..23; lane>=16 ->
      // K 8..15,24..31 within this 32-wide chunk.
      AFrag af;
      const int row = mstrip * 16 + (lane & 15);
      const int kb = ((lane < 16) ? 0 : 8) + (c << 5);
      af.u[0] = *(const uint4*)&Ast[row * 64 + kb];
      af.u[1] = *(const uint4*)&Ast[row * 64 + kb + 16];
#pragma unroll
      for (int j = 0; j < TILES; ++j) {
        // B frag (ISA 16-bit B 32x16): lane<16 -> K 0..7,16..23 col=lane; lane>=16
        // -> K 8..15,24..31 col=lane-16. BT is K-contiguous per column -> 2x b128
        // straight from global (L2-resident).
        const int n = (nbase + j) * 16 + (lane & 15);
        const int kbb = (s << 6) + (c << 5) + ((lane < 16) ? 0 : 8);
        const unsigned short* bp = BT + (size_t)n * Ktot + kbb;
        AFrag bf;
        bf.u[0] = *(const uint4*)bp;
        bf.u[1] = *(const uint4*)(bp + 16);
        acc[j] = __builtin_amdgcn_wmma_f32_16x16x32_bf16(
            false, af.v, false, bf.v, (short)0, acc[j], false, false);
      }
    }
    // Prefetch stage s+7 into ring slot (s+7)&7 == (s-1)&7; its last reader
    // was compute(s-1), separated from this issue by iter s-1's barrier.
    // s_wait_asynccnt 12: stage = 2 in-order ops, 6 newer stages may remain
    // outstanding -> stage s+1 is resident before the barrier.
    if (s + 7 < steps) { issueA(s + 7); wait_async12(); }
    else               { wait_async0(); }
    __syncthreads();
  }

  // Epilogue. f32 C layout: col = lane&15, VGPR v -> row v (lane<16) / 8+v.
  const int col = lane & 15;
  const int rbase = m0 + mstrip * 16 + ((lane < 16) ? 0 : 8);
  float svv[8];
  if (EPI == 1 || EPI == 2 || EPI == 3 || EPI == 4) {
#pragma unroll
    for (int v = 0; v < 8; ++v) svv[v] = scale[rbase + v];
  }
#pragma unroll
  for (int j = 0; j < TILES; ++j) {
    const int n = (nbase + j) * 16 + col;
    const float bv = (EPI == 0 || EPI == 1 || EPI == 4) ? bias[n] : 0.0f;
    unsigned short hv[8];
#pragma unroll
    for (int v = 0; v < 8; ++v) {
      const int r = rbase + v;
      float x = acc[j][v];
      if (EPI == 0) {
        x = fmaxf(x + bv, 0.0f);
        out0[(size_t)r * NCOLS + n] = f32_to_bf16(x);
      } else if (EPI == 1) {
        x = fmaxf(x + bv, 0.0f);
        out0[(size_t)r * NCOLS + n] = f32_to_bf16(x);
        hv[v] = f32_to_bf16(svv[v] * x);
      } else if (EPI == 2) {
        hv[v] = f32_to_bf16(svv[v] * svv[v] * x);
      } else if (EPI == 3) {
        out0[(size_t)r * NCOLS + n] = f32_to_bf16(svv[v] * x);
      } else {
        outf[(size_t)r * NCOLS + n] = (x + bv) * svv[v];
      }
    }
    if (EPI == 1 || EPI == 2) {   // packed transposed store: 8 rows, 1 col
      *(uint4*)&outT[(size_t)n * Mrows + rbase] = *(uint4*)hv;
    }
  }
}

// ------------------------------- driver ------------------------------------

extern "C" void kernel_launch(void* const* d_in, const int* in_sizes, int n_in,
                              void* d_out, int out_size, void* d_ws, size_t ws_size,
                              hipStream_t stream) {
  (void)in_sizes; (void)n_in; (void)out_size; (void)ws_size;
  const int N = 8192;

  const float* features  = (const float*)d_in[0];
  const float* adjacency = (const float*)d_in[1];
  const float* mask      = (const float*)d_in[2];
  const float* W_e1  = (const float*)d_in[3];  const float* b_e1  = (const float*)d_in[4];
  const float* W_e2  = (const float*)d_in[5];  const float* b_e2  = (const float*)d_in[6];
  const float* W_gcn = (const float*)d_in[7];  const float* b_gcn = (const float*)d_in[8];
  const float* W_gd  = (const float*)d_in[9];  const float* b_gd  = (const float*)d_in[10];
  const float* W_p1  = (const float*)d_in[11]; const float* b_p1  = (const float*)d_in[12];
  const float* W_p2  = (const float*)d_in[13]; const float* b_p2  = (const float*)d_in[14];
  const float* W_pi  = (const float*)d_in[15]; const float* b_pi  = (const float*)d_in[16];

  char* ws = (char*)d_ws;
  size_t off = 0;
  auto alloc = [&](size_t bytes) -> void* {
    void* p = ws + off;
    off = (off + bytes + 255) & ~(size_t)255;
    return p;
  };

  unsigned short* Bt    = (unsigned short*)alloc((size_t)N * N * 2);  // bf16 B^T
  float*          deg   = (float*)alloc((size_t)N * 4);
  float*          dinv  = (float*)alloc((size_t)N * 4);
  unsigned short* Fb    = (unsigned short*)alloc((size_t)N * 64 * 2);
  unsigned short* We1t  = (unsigned short*)alloc(64 * 64 * 2);        // [64][64]
  unsigned short* We2t  = (unsigned short*)alloc(64 * 256 * 2);       // [256][64]
  unsigned short* Wgcnt = (unsigned short*)alloc(256 * 256 * 2);      // [256][256]
  unsigned short* Wgdt  = (unsigned short*)alloc(256 * 256 * 2);      // [256][256]
  unsigned short* Wp1t  = (unsigned short*)alloc(512 * 128 * 2);      // [128][512]
  unsigned short* Wp2t  = (unsigned short*)alloc(128 * 128 * 2);      // [128][128]
  unsigned short* Wpit  = (unsigned short*)alloc(128 * 32 * 2);       // [32][128]
  unsigned short* X1    = (unsigned short*)alloc((size_t)N * 64 * 2);
  unsigned short* Xb    = (unsigned short*)alloc((size_t)N * 256 * 2);
  unsigned short* V1t   = (unsigned short*)alloc((size_t)N * 256 * 2); // [256][N]
  unsigned short* V2t   = (unsigned short*)alloc((size_t)N * 256 * 2); // [256][N]
  unsigned short* Hb    = (unsigned short*)alloc((size_t)N * 256 * 2);
  unsigned short* Xg1   = (unsigned short*)alloc((size_t)N * 256 * 2);
  unsigned short* Xg    = (unsigned short*)alloc((size_t)N * 256 * 2);
  unsigned short* P1    = (unsigned short*)alloc((size_t)N * 128 * 2);
  unsigned short* P2    = (unsigned short*)alloc((size_t)N * 128 * 2);

  // adjacency pass: transpose->bf16 with fused degree counts, then dinv
  zero_f32_kernel<<<N / 256, 256, 0, stream>>>(deg, N);
  transpose_deg_kernel<<<dim3(N / 64, N / 64), 256, 0, stream>>>(adjacency, Bt, deg, N);
  dinv_kernel<<<N / 256, 256, 0, stream>>>(deg, dinv, N);

  // operand conversions
  cvt_f32_to_bf16_kernel<<<(N * 64 + 255) / 256, 256, 0, stream>>>(features, Fb, N * 64);
  auto cvtT = [&](const float* s, unsigned short* d, int K, int Nc) {
    cvt_transpose_bf16_kernel<<<(K * Nc + 255) / 256, 256, 0, stream>>>(s, d, K, Nc);
  };
  cvtT(W_e1, We1t, 64, 64);
  cvtT(W_e2, We2t, 64, 256);
  cvtT(W_gcn, Wgcnt, 256, 256);
  cvtT(W_gd, Wgdt, 256, 256);
  cvtT(W_p1, Wp1t, 512, 128);
  cvtT(W_p2, Wp2t, 128, 128);
  cvtT(W_pi, Wpit, 128, 32);

  const dim3 gg(N / 64);  // 128 blocks, M=8192 for every GEMM

  // 1) X1 = relu(F @ W_e1 + b_e1)                        [N,64]
  gemm_bf16_wmma<64, 0><<<gg, 256, 0, stream>>>(Fb, 64, nullptr, 0, 64,
      We1t, b_e1, nullptr, X1, nullptr, nullptr, N);
  // 2) X = relu(X1 @ W_e2 + b_e2); V1t = (dinv*X)^T      [N,256]
  gemm_bf16_wmma<256, 1><<<gg, 256, 0, stream>>>(X1, 64, nullptr, 0, 64,
      We2t, b_e2, dinv, Xb, V1t, nullptr, N);
  // 3) V2t = (dinv^2 * (B^T @ V1))^T                     [N,256]
  gemm_bf16_wmma<256, 2><<<gg, 256, 0, stream>>>(Bt, N, nullptr, 0, N,
      V1t, nullptr, dinv, nullptr, V2t, nullptr, N);
  // 4) h = dinv * (B^T @ V2)      (== prop(prop(X)))     [N,256]
  gemm_bf16_wmma<256, 3><<<gg, 256, 0, stream>>>(Bt, N, nullptr, 0, N,
      V2t, nullptr, dinv, Hb, nullptr, nullptr, N);
  // 5) Xg1 = relu(h @ W_gcn + b_gcn)                     [N,256]
  gemm_bf16_wmma<256, 0><<<gg, 256, 0, stream>>>(Hb, 256, nullptr, 0, 256,
      Wgcnt, b_gcn, nullptr, Xg1, nullptr, nullptr, N);
  // 6) Xg = relu(Xg1 @ W_gd + b_gd)                      [N,256]
  gemm_bf16_wmma<256, 0><<<gg, 256, 0, stream>>>(Xg1, 256, nullptr, 0, 256,
      Wgdt, b_gd, nullptr, Xg, nullptr, nullptr, N);
  // 7) P1 = relu([Xg|X] @ W_p1 + b_p1)  (two K phases)   [N,128]
  gemm_bf16_wmma<128, 0><<<gg, 256, 0, stream>>>(Xg, 256, Xb, 256, 256,
      Wp1t, b_p1, nullptr, P1, nullptr, nullptr, N);
  // 8) P2 = relu(P1 @ W_p2 + b_p2)                       [N,128]
  gemm_bf16_wmma<128, 0><<<gg, 256, 0, stream>>>(P1, 128, nullptr, 0, 128,
      Wp2t, b_p2, nullptr, P2, nullptr, nullptr, N);
  // 9) pi = (P2 @ W_pi + b_pi) * mask  -> f32 d_out      [N,32]
  gemm_bf16_wmma<32, 4><<<gg, 256, 0, stream>>>(P2, 128, nullptr, 0, 128,
      Wpit, b_pi, mask, nullptr, nullptr, (float*)d_out, N);
}
